// PointerNet_21303037788550
// MI455X (gfx1250) — matrix operand order repaired and checked by reading, not compiled, not run on hardware
//
#include <hip/hip_runtime.h>
#include <cstdint>
#include <cstddef>

// Problem constants (match reference)
#define B_    256
#define SEQ_  256
#define EMB_  128
#define HID_  512
#define G4H_  2048          // 4*HID
#define NEGV  (-1e9f)
#define CEXPV (10.0f)

typedef __bf16 bf16;
typedef __attribute__((ext_vector_type(16))) __bf16 v16bf;
typedef __attribute__((ext_vector_type(8)))  float  v8f;

__device__ __forceinline__ float bf2f(uint32_t lo16) {
  uint32_t u = lo16 << 16;
  return __builtin_bit_cast(float, u);
}

// ---------------------------------------------------------------------------
// gfx1250 async copy to LDS (CDNA5 ISA): per-lane
// global->LDS DMA tracked by ASYNCcnt. Generic pointers to __shared__ carry
// the LDS byte offset in their low 32 bits (aperture rules, 00_overview §10.2).
// ---------------------------------------------------------------------------
__device__ __forceinline__ uint32_t lds_off_of(const void* p) {
  return (uint32_t)(uintptr_t)p;
}

// Stage one 1KB ref row (512 bf16) into LDS: two b128 issues per lane.
__device__ __forceinline__ void async_copy_row(uint32_t lds_base, const bf16* grow, int lane) {
  const char* g = (const char*)grow + lane * 16;
  asm volatile("global_load_async_to_lds_b128 %0, %1, off"
               :: "v"(lds_base + lane * 16), "v"(g) : "memory");
  asm volatile("global_load_async_to_lds_b128 %0, %1, off"
               :: "v"(lds_base + 512 + lane * 16), "v"(g + 512) : "memory");
}

// ---------------------------------------------------------------------------
// WMMA fragment loaders (CDNA5 16x16x32 bf16 layouts, CDNA5 ISA)
//   A (16x32, MxK): lane m = lane&15. lanes 0-15 hold K[0..7] then K[16..23],
//                   lanes 16-31 hold K[8..15] then K[24..31].
//   B (32x16, KxN): lane n = lane&15. lanes 0-15 hold K[0..7] then K[16..23],
//                   lanes 16-31 hold K[8..15] then K[24..31].
// Both map to contiguous 16B loads from row-major A[M,K] and W[N,K] (B = Wᵀ).
// ---------------------------------------------------------------------------
struct B32frag { uint4 a, b; };

__device__ __forceinline__ v16bf ld2(const bf16* p0, const bf16* p1) {
  B32frag t;
  t.a = *(const uint4*)p0;
  t.b = *(const uint4*)p1;
  return __builtin_bit_cast(v16bf, t);
}

__device__ __forceinline__ v16bf loadA(const bf16* A, int lda, int m0, int kb, int lane) {
  int row = m0 + (lane & 15);
  int kh  = (lane >> 4) << 3;                       // 0 or 8
  const bf16* p = A + (size_t)row * lda + kb + kh;
  return ld2(p, p + 16);
}

__device__ __forceinline__ v16bf loadB(const bf16* W, int ldw, int n0, int kb, int lane) {
  int col = n0 + (lane & 15);
  int kh  = (lane >> 4) << 3;                       // 0 or 8
  const bf16* p = W + (size_t)col * ldw + kb + kh;
  return ld2(p, p + 16);
}

__device__ __forceinline__ v8f wmma_bf16(v16bf a, v16bf b, v8f c) {
  return __builtin_amdgcn_wmma_f32_16x16x32_bf16(false, a, false, b, (short)0, c, false, false);
}

// ---------------------------------------------------------------------------
// Fused dual-K GEMM:  C[M,N] = A1[M,K1]@B1ᵀ + A2[M,K2]@B2ᵀ + bias.
// One wave owns a 16x64 output block: 4 independent accumulators break the
// WMMA->WMMA RAW chain and reuse each A fragment 4x. BF16OUT selects the
// output precision (bf16 for the L2-resident attention-ref tensors).
// ---------------------------------------------------------------------------
template <bool BF16OUT>
__global__ __launch_bounds__(256) void gemm2_kernel(
    const bf16* __restrict__ A1, int lda1, const bf16* __restrict__ B1, int ldb1, int K1,
    const bf16* __restrict__ A2, int lda2, const bf16* __restrict__ B2, int ldb2, int K2,
    const float* __restrict__ bias, void* __restrict__ Cout, int ldc, int M, int N)
{
  int lane  = threadIdx.x & 31;
  int wave  = threadIdx.x >> 5;
  int wtile = blockIdx.x * (blockDim.x >> 5) + wave;
  int tilesN = N >> 6;                              // 64-wide N blocks
  int tilesT = (M >> 4) * tilesN;
  if (wtile >= tilesT) return;                      // wave-uniform: EXEC stays full
  int m0 = (wtile / tilesN) << 4;
  int n0 = (wtile % tilesN) << 6;

  v8f acc[4] = {};
  for (int kb = 0; kb < K1; kb += 32) {
    v16bf a  = loadA(A1, lda1, m0, kb, lane);
    v16bf b0 = loadB(B1, ldb1, n0,      kb, lane);
    v16bf b1 = loadB(B1, ldb1, n0 + 16, kb, lane);
    v16bf b2 = loadB(B1, ldb1, n0 + 32, kb, lane);
    v16bf b3 = loadB(B1, ldb1, n0 + 48, kb, lane);
    acc[0] = wmma_bf16(a, b0, acc[0]);
    acc[1] = wmma_bf16(a, b1, acc[1]);
    acc[2] = wmma_bf16(a, b2, acc[2]);
    acc[3] = wmma_bf16(a, b3, acc[3]);
  }
  if (K2 > 0) {
    for (int kb = 0; kb < K2; kb += 32) {
      v16bf a  = loadA(A2, lda2, m0, kb, lane);
      v16bf b0 = loadB(B2, ldb2, n0,      kb, lane);
      v16bf b1 = loadB(B2, ldb2, n0 + 16, kb, lane);
      v16bf b2 = loadB(B2, ldb2, n0 + 32, kb, lane);
      v16bf b3 = loadB(B2, ldb2, n0 + 48, kb, lane);
      acc[0] = wmma_bf16(a, b0, acc[0]);
      acc[1] = wmma_bf16(a, b1, acc[1]);
      acc[2] = wmma_bf16(a, b2, acc[2]);
      acc[3] = wmma_bf16(a, b3, acc[3]);
    }
  }

  // C/D layout: VGPR r -> M = m0 + r + (lane>=16 ? 8 : 0), N = n0 + (lane&15)
  int nn = lane & 15;
  int mb = m0 + ((lane >> 4) << 3);
#pragma unroll
  for (int j = 0; j < 4; ++j) {
    int n = n0 + 16 * j + nn;
    float bv = bias ? bias[n] : 0.0f;
#pragma unroll
    for (int r = 0; r < 8; ++r) {
      float v = acc[j][r] + bv;
      if (BF16OUT) ((bf16*)Cout)[(size_t)(mb + r) * ldc + n] = (bf16)v;
      else         ((float*)Cout)[(size_t)(mb + r) * ldc + n] = v;
    }
  }
}

// ---------------------------------------------------------------------------
// Elementwise / small kernels
// ---------------------------------------------------------------------------
__device__ __forceinline__ float sigm(float x) { return 1.0f / (1.0f + __expf(-x)); }

__global__ __launch_bounds__(256) void lstm_gate_kernel(
    const float* __restrict__ g, float* __restrict__ c, bf16* __restrict__ h,
    bf16* __restrict__ enc_out, int t)
{
  int idx = blockIdx.x * blockDim.x + threadIdx.x;  // b*HID + j, total B*HID
  int b = idx >> 9, j = idx & (HID_ - 1);
  const float* gb = g + (size_t)b * G4H_;
  float iv = gb[j], fv = gb[HID_ + j], gv = gb[2 * HID_ + j], ov = gb[3 * HID_ + j];
  float cv = sigm(fv) * c[idx] + sigm(iv) * tanhf(gv);
  float hv = sigm(ov) * tanhf(cv);
  c[idx] = cv;
  h[idx] = (bf16)hv;
  if (enc_out) enc_out[((size_t)b * SEQ_ + t) * HID_ + j] = (bf16)hv;
}

__global__ __launch_bounds__(256) void gather_emb_kernel(
    const float* __restrict__ embedding, const int* __restrict__ inputs, bf16* __restrict__ emb)
{
  int idx = blockIdx.x * blockDim.x + threadIdx.x;  // over B*S*E
  int e  = idx & (EMB_ - 1);
  int bs = idx >> 7;
  int tok = inputs[bs];
  emb[idx] = (bf16)embedding[(size_t)tok * EMB_ + e];
}

__global__ __launch_bounds__(256) void cvt_bf16_kernel(const float* __restrict__ s, bf16* __restrict__ d, int n)
{
  int i = blockIdx.x * blockDim.x + threadIdx.x;
  if (i < n) d[i] = (bf16)s[i];
}

__global__ __launch_bounds__(256) void zero_u32_kernel(uint32_t* __restrict__ p, int n)
{
  int i = blockIdx.x * blockDim.x + threadIdx.x;
  if (i < n) p[i] = 0u;
}

__global__ __launch_bounds__(256) void init_decin_kernel(const float* __restrict__ ds, bf16* __restrict__ dec_in)
{
  int idx = blockIdx.x * blockDim.x + threadIdx.x;  // B*EMB
  dec_in[idx] = (bf16)ds[idx & (EMB_ - 1)];
}

// Deterministic hash-based uniform for Gumbel noise (JAX bits unreproducible).
__device__ __forceinline__ float u01(uint32_t a, uint32_t b, uint32_t c)
{
  uint32_t x = a * 0x9E3779B9u ^ b * 0x85EBCA6Bu ^ c * 0xC2B2AE35u;
  x ^= x >> 16; x *= 0x7FEB352Du;
  x ^= x >> 15; x *= 0x846CA68Bu;
  x ^= x >> 16;
  return (float)(x >> 8) * (1.0f / 16777216.0f);
}

// ---------------------------------------------------------------------------
// Attention score pass with wave-private async double buffering.
// Wave wv walks rows s = wv, wv+8, ...; while computing tanh reductions on the
// current 1KB row staged in its own LDS slot, the next row streams in via
// GLOBAL_LOAD_ASYNC_TO_LDS_B128. ASYNCcnt<=2 releases exactly the current row
// (async loads complete in issue order); no barriers needed (wave-private).
// ---------------------------------------------------------------------------
__device__ __forceinline__ void attn_scores_async(
    const bf16* __restrict__ refb,   // ref + b*SEQ*HID
    const float* __restrict__ sq, const float* __restrict__ sv,
    const float* __restrict__ maskb, float* __restrict__ sout,
    bf16* __restrict__ rbuf,         // this wave's [2][512] LDS slot
    int lane, int wv, float cexp)
{
  uint32_t lbase = lds_off_of(rbuf);
  async_copy_row(lbase, refb + (size_t)wv * HID_, lane);
  for (int i = 0; i < 32; ++i) {
    int s   = wv + i * 8;
    int nxt = s + 8;
    if (nxt < SEQ_) {
      async_copy_row(lbase + ((i + 1) & 1) * 1024, refb + (size_t)nxt * HID_, lane);
      asm volatile("s_wait_asynccnt 0x2" ::: "memory");
    } else {
      asm volatile("s_wait_asynccnt 0x0" ::: "memory");
    }
    const uint32_t* rp = (const uint32_t*)(rbuf + (i & 1) * 512);
    float acc = 0.0f;
#pragma unroll
    for (int it = 0; it < 8; ++it) {
      uint32_t pk = rp[lane + 32 * it];
      int h = 2 * (lane + 32 * it);
      acc += tanhf(sq[h]     + bf2f(pk & 0xffffu)) * sv[h];
      acc += tanhf(sq[h + 1] + bf2f(pk >> 16))     * sv[h + 1];
    }
#pragma unroll
    for (int o = 16; o; o >>= 1) acc += __shfl_xor(acc, o, 32);
    if (lane == 0) {
      float v = (cexp > 0.0f) ? cexp * tanhf(acc) : acc;
      sout[s] = (maskb[s] != 0.0f) ? NEGV : v;
    }
  }
}

// ---------------------------------------------------------------------------
// Fused glimpse: scores = tanh(q + ref)·V (masked) -> softmax -> q' = Σ w·ref.
// One block per batch row; q and V staged in LDS once; ref tensors are bf16 so
// both attention refs (134 MB total) stay resident in the 192 MB L2 across
// all 256 decoder steps.
// ---------------------------------------------------------------------------
__global__ __launch_bounds__(256) void glimpse_kernel(
    const float* __restrict__ qg,   // [B,H] f32
    const bf16* __restrict__ ref,   // [B,S,H] bf16
    const float* __restrict__ V,    // [H]
    const float* __restrict__ mask, // [B,S]
    bf16* __restrict__ qout)        // [B,H] bf16
{
  __shared__ float sq[HID_];
  __shared__ float sv[HID_];
  __shared__ float sc[SEQ_];
  __shared__ bf16  rbuf[8][2][HID_];   // per-wave async double buffers (16 KB)
  int b = blockIdx.x, tid = threadIdx.x;
  int lane = tid & 31, wv = tid >> 5;

  sq[tid]       = qg[(size_t)b * HID_ + tid];
  sq[tid + 256] = qg[(size_t)b * HID_ + tid + 256];
  sv[tid]       = V[tid];
  sv[tid + 256] = V[tid + 256];
  __syncthreads();

  attn_scores_async(ref + (size_t)b * SEQ_ * HID_, sq, sv,
                    mask + (size_t)b * SEQ_, sc, &rbuf[wv][0][0], lane, wv, 0.0f);
  __syncthreads();

  // Softmax over sc (reuse sq[0..255] as reduction scratch)
  float v = sc[tid];
  sq[tid] = v; __syncthreads();
  for (int o = 128; o; o >>= 1) { if (tid < o) sq[tid] = fmaxf(sq[tid], sq[tid + o]); __syncthreads(); }
  float mx = sq[0]; __syncthreads();
  float e = __expf(v - mx);
  sq[tid] = e; __syncthreads();
  for (int o = 128; o; o >>= 1) { if (tid < o) sq[tid] += sq[tid + o]; __syncthreads(); }
  sc[tid] = e / sq[0];
  __syncthreads();

  // Weighted sum over s (thread covers h=tid and h=tid+256), prefetch ahead
  const uint16_t* rb = (const uint16_t*)(ref + (size_t)b * SEQ_ * HID_);
  float a0 = 0.0f, a1 = 0.0f;
  for (int s = 0; s < SEQ_; ++s) {
    if (lane == 0 && s + 8 < SEQ_)
      __builtin_prefetch(rb + (size_t)(s + 8) * HID_ + tid, 0, 0);   // global_prefetch_b8
    float wv_ = sc[s];
    a0 += wv_ * bf2f(rb[(size_t)s * HID_ + tid]);
    a1 += wv_ * bf2f(rb[(size_t)s * HID_ + tid + 256]);
  }
  qout[(size_t)b * HID_ + tid]       = (bf16)a0;
  qout[(size_t)b * HID_ + tid + 256] = (bf16)a1;
}

// ---------------------------------------------------------------------------
// Fused pointer + sampling: lg = CEXP*tanh(tanh(q + ref)·V) (masked), then
// Gumbel-max argmax, log-softmax at the chosen index, mask update, and gather
// of the chosen city's embedding as the next decoder input.
// ---------------------------------------------------------------------------
__global__ __launch_bounds__(256) void pointer_sample_kernel(
    const float* __restrict__ qp,   // [B,H]
    const bf16* __restrict__ ref,   // [B,S,H] bf16
    const float* __restrict__ V,
    float* __restrict__ mask,
    const bf16* __restrict__ emb, bf16* __restrict__ dec_in,
    float* __restrict__ out_tour, float* __restrict__ out_logp, int t)
{
  __shared__ float sq[HID_];
  __shared__ float sv[HID_];
  __shared__ float sl[SEQ_];
  __shared__ float skey[SEQ_];
  __shared__ int   sidx[SEQ_];
  __shared__ int   sptr;
  __shared__ bf16  rbuf[8][2][HID_];
  int b = blockIdx.x, tid = threadIdx.x;
  int lane = tid & 31, wv = tid >> 5;

  sq[tid]       = qp[(size_t)b * HID_ + tid];
  sq[tid + 256] = qp[(size_t)b * HID_ + tid + 256];
  sv[tid]       = V[tid];
  sv[tid + 256] = V[tid + 256];
  __syncthreads();

  attn_scores_async(ref + (size_t)b * SEQ_ * HID_, sq, sv,
                    mask + (size_t)b * SEQ_, sl, &rbuf[wv][0][0], lane, wv, CEXPV);
  __syncthreads();

  // Gumbel-max argmax
  float l = sl[tid];
  float u = u01((uint32_t)t, (uint32_t)b, (uint32_t)tid);
  float gum = -logf(-logf(u + 1e-10f) + 1e-10f);
  skey[tid] = l + gum; sidx[tid] = tid;
  __syncthreads();
  for (int o = 128; o; o >>= 1) {
    if (tid < o && skey[tid + o] > skey[tid]) { skey[tid] = skey[tid + o]; sidx[tid] = sidx[tid + o]; }
    __syncthreads();
  }
  if (tid == 0) sptr = sidx[0];
  __syncthreads();
  int ptr = sptr;

  // log-softmax denominator (reuse sq[0..255])
  sq[tid] = l; __syncthreads();
  for (int o = 128; o; o >>= 1) { if (tid < o) sq[tid] = fmaxf(sq[tid], sq[tid + o]); __syncthreads(); }
  float mx = sq[0]; __syncthreads();
  sq[tid] = __expf(l - mx); __syncthreads();
  for (int o = 128; o; o >>= 1) { if (tid < o) sq[tid] += sq[tid + o]; __syncthreads(); }

  if (tid == 0) {
    float lse = mx + logf(sq[0]);
    out_tour[(size_t)b * SEQ_ + t] = (float)ptr;
    out_logp[(size_t)b * SEQ_ + t] = sl[ptr] - lse;
    mask[(size_t)b * SEQ_ + ptr] = 1.0f;
  }
  if (tid < EMB_) dec_in[(size_t)b * EMB_ + tid] = emb[((size_t)b * SEQ_ + ptr) * EMB_ + tid];
}

// ---------------------------------------------------------------------------
// Host orchestration
// ---------------------------------------------------------------------------
extern "C" void kernel_launch(void* const* d_in, const int* in_sizes, int n_in,
                              void* d_out, int out_size, void* d_ws, size_t ws_size,
                              hipStream_t stream)
{
  (void)in_sizes; (void)n_in; (void)out_size; (void)ws_size;

  const float* embedding = (const float*)d_in[0];
  const float* enc_Wih = (const float*)d_in[1];
  const float* enc_Whh = (const float*)d_in[2];
  const float* enc_b   = (const float*)d_in[3];
  const float* dec_Wih = (const float*)d_in[4];
  const float* dec_Whh = (const float*)d_in[5];
  const float* dec_b   = (const float*)d_in[6];
  const float* pt_Wq   = (const float*)d_in[7];
  const float* pt_bq   = (const float*)d_in[8];
  const float* pt_Wref = (const float*)d_in[9];
  const float* pt_bref = (const float*)d_in[10];
  const float* pt_V    = (const float*)d_in[11];
  const float* gl_Wq   = (const float*)d_in[12];
  const float* gl_bq   = (const float*)d_in[13];
  const float* gl_Wref = (const float*)d_in[14];
  const float* gl_bref = (const float*)d_in[15];
  const float* gl_V    = (const float*)d_in[16];
  const float* dec_st  = (const float*)d_in[17];
  const int*   inputs  = (const int*)d_in[18];

  float* out_tour = (float*)d_out;
  float* out_logp = out_tour + (size_t)B_ * SEQ_;

  // -------- workspace layout --------
  char* base = (char*)d_ws;
  size_t off = 0;
  auto alloc = [&](size_t bytes) -> void* {
    void* p = base + off;
    off += (bytes + 255) & ~(size_t)255;
    return p;
  };
  bf16*  emb_bf     = (bf16*) alloc((size_t)B_ * SEQ_ * EMB_ * 2);
  bf16*  enc_out_bf = (bf16*) alloc((size_t)B_ * SEQ_ * HID_ * 2);
  bf16*  gl_ref     = (bf16*) alloc((size_t)B_ * SEQ_ * HID_ * 2);   // bf16: L2-resident
  bf16*  pt_ref     = (bf16*) alloc((size_t)B_ * SEQ_ * HID_ * 2);   // bf16: L2-resident
  bf16*  eWih       = (bf16*) alloc((size_t)G4H_ * EMB_ * 2);
  bf16*  eWhh       = (bf16*) alloc((size_t)G4H_ * HID_ * 2);
  bf16*  dWih       = (bf16*) alloc((size_t)G4H_ * EMB_ * 2);
  bf16*  dWhh       = (bf16*) alloc((size_t)G4H_ * HID_ * 2);
  bf16*  gWq        = (bf16*) alloc((size_t)HID_ * HID_ * 2);
  bf16*  pWq        = (bf16*) alloc((size_t)HID_ * HID_ * 2);
  bf16*  gWr        = (bf16*) alloc((size_t)HID_ * HID_ * 2);
  bf16*  pWr        = (bf16*) alloc((size_t)HID_ * HID_ * 2);
  bf16*  h_bf       = (bf16*) alloc((size_t)B_ * HID_ * 2);
  float* c_st       = (float*)alloc((size_t)B_ * HID_ * 4);
  float* g_buf      = (float*)alloc((size_t)B_ * G4H_ * 4);
  float* qg         = (float*)alloc((size_t)B_ * HID_ * 4);
  float* qp         = (float*)alloc((size_t)B_ * HID_ * 4);
  float* mask       = (float*)alloc((size_t)B_ * SEQ_ * 4);
  bf16*  q_bf       = (bf16*) alloc((size_t)B_ * HID_ * 2);
  bf16*  dec_in     = (bf16*) alloc((size_t)B_ * EMB_ * 2);

  auto gemm2f = [&](const bf16* A1, int lda1, const bf16* B1, int ldb1, int K1,
                    const bf16* A2, int lda2, const bf16* B2, int ldb2, int K2,
                    const float* bias, float* C, int ldc, int M, int N) {
    int waves  = (M >> 4) * (N >> 6);
    int blocks = (waves + 7) / 8;
    gemm2_kernel<false><<<blocks, 256, 0, stream>>>(A1, lda1, B1, ldb1, K1,
                                                    A2, lda2, B2, ldb2, K2, bias, C, ldc, M, N);
  };
  auto gemm2b = [&](const bf16* A1, int lda1, const bf16* B1, int ldb1, int K1,
                    const float* bias, bf16* C, int ldc, int M, int N) {
    int waves  = (M >> 4) * (N >> 6);
    int blocks = (waves + 7) / 8;
    gemm2_kernel<true><<<blocks, 256, 0, stream>>>(A1, lda1, B1, ldb1, K1,
                                                   nullptr, 0, nullptr, 0, 0, bias, C, ldc, M, N);
  };
  auto cvt = [&](const float* s, bf16* d, int n) {
    cvt_bf16_kernel<<<(n + 255) / 256, 256, 0, stream>>>(s, d, n);
  };

  // -------- prep: weight conversion, state init, embedding gather --------
  cvt(enc_Wih, eWih, G4H_ * EMB_);
  cvt(enc_Whh, eWhh, G4H_ * HID_);
  cvt(dec_Wih, dWih, G4H_ * EMB_);
  cvt(dec_Whh, dWhh, G4H_ * HID_);
  cvt(gl_Wq, gWq, HID_ * HID_);
  cvt(pt_Wq, pWq, HID_ * HID_);
  cvt(gl_Wref, gWr, HID_ * HID_);
  cvt(pt_Wref, pWr, HID_ * HID_);

  zero_u32_kernel<<<(B_ * HID_ / 2 + 255) / 256, 256, 0, stream>>>((uint32_t*)h_bf, B_ * HID_ / 2);
  zero_u32_kernel<<<(B_ * HID_ + 255) / 256, 256, 0, stream>>>((uint32_t*)c_st, B_ * HID_);
  zero_u32_kernel<<<(B_ * SEQ_ + 255) / 256, 256, 0, stream>>>((uint32_t*)mask, B_ * SEQ_);
  init_decin_kernel<<<B_ * EMB_ / 256, 256, 0, stream>>>(dec_st, dec_in);
  gather_emb_kernel<<<B_ * SEQ_ * EMB_ / 256, 256, 0, stream>>>(embedding, inputs, emb_bf);

  // -------- encoder: 256 sequential LSTM steps --------
  for (int t = 0; t < SEQ_; ++t) {
    gemm2f(emb_bf + (size_t)t * EMB_, SEQ_ * EMB_, eWih, EMB_, EMB_,
           h_bf, HID_, eWhh, HID_, HID_,
           enc_b, g_buf, G4H_, B_, G4H_);
    lstm_gate_kernel<<<B_ * HID_ / 256, 256, 0, stream>>>(g_buf, c_st, h_bf, enc_out_bf, t);
  }

  // -------- attention reference projections (big WMMA GEMMs, bf16 out) ----
  gemm2b(enc_out_bf, HID_, gWr, HID_, HID_, gl_bref, gl_ref, HID_, B_ * SEQ_, HID_);
  gemm2b(enc_out_bf, HID_, pWr, HID_, HID_, pt_bref, pt_ref, HID_, B_ * SEQ_, HID_);

  // -------- decoder: 256 sequential pointer steps --------
  for (int t = 0; t < SEQ_; ++t) {
    // LSTM cell
    gemm2f(dec_in, EMB_, dWih, EMB_, EMB_,
           h_bf, HID_, dWhh, HID_, HID_,
           dec_b, g_buf, G4H_, B_, G4H_);
    lstm_gate_kernel<<<B_ * HID_ / 256, 256, 0, stream>>>(g_buf, c_st, h_bf, nullptr, 0);

    // glimpse (fused score -> softmax -> weighted sum)
    gemm2f(h_bf, HID_, gWq, HID_, HID_, nullptr, 0, nullptr, 0, 0,
           gl_bq, qg, HID_, B_, HID_);
    glimpse_kernel<<<B_, 256, 0, stream>>>(qg, gl_ref, gl_V, mask, q_bf);

    // pointer (fused score -> Gumbel sample -> logp -> mask -> next input)
    gemm2f(q_bf, HID_, pWq, HID_, HID_, nullptr, 0, nullptr, 0, 0,
           pt_bq, qp, HID_, B_, HID_);
    pointer_sample_kernel<<<B_, 256, 0, stream>>>(qp, pt_ref, pt_V, mask,
                                                  emb_bf, dec_in, out_tour, out_logp, t);
  }
}
